// SimpleRNN_40200893890811
// MI455X (gfx1250) — matrix operand order repaired and checked by reading, not compiled, not run on hardware
//
#include <hip/hip_runtime.h>
#include <hip/hip_bf16.h>

typedef __bf16 bf16_t;
typedef bf16_t v16bf __attribute__((ext_vector_type(16)));
typedef float  v8f   __attribute__((ext_vector_type(8)));
typedef unsigned short u16;
typedef u16 v8u16 __attribute__((ext_vector_type(8)));

union BFrag {
    v16bf v;
    v8u16 h[2];
    u16   s[16];
};

// float -> bf16 bits, round-to-nearest-even
__device__ __forceinline__ u16 f2bfbits(float f) {
    union { float f; unsigned u; } c; c.f = f;
    unsigned r = c.u + 0x7FFFu + ((c.u >> 16) & 1u);
    return (u16)(r >> 16);
}

__device__ __forceinline__ v8f wmma_bf16(const BFrag& a, const BFrag& b, v8f c) {
    return __builtin_amdgcn_wmma_f32_16x16x32_bf16(false, a.v, false, b.v,
                                                   (short)0, c, false, false);
}

#define T_DIM 1024
#define B_DIM 64
#define U_DIM 256
#define C_DIM 64
#define O_DIM 20

// ---------------------------------------------------------------------------
// Kernel 1: BN1(x) then xproj = xn @ Wx + bh    (M=65536, K=64, N=256)
// 512 WGs x 256 threads; each WG does 128 (b,t) rows x 256 cols.
// Output layout: xproj[t][b][u]  (scan-friendly: one step = contiguous 64 KB)
// ---------------------------------------------------------------------------
__global__ __launch_bounds__(256) void k_bn1_xproj(
    const float* __restrict__ x,
    const float* __restrict__ g1, const float* __restrict__ b1,
    const float* __restrict__ m1, const float* __restrict__ v1,
    const float* __restrict__ Wx, const float* __restrict__ bh,
    float* __restrict__ xproj)
{
    __shared__ __align__(16) u16 lA[128 * C_DIM];       // bn1(x) tile, bf16
    __shared__ __align__(16) u16 lWxT[U_DIM * C_DIM];   // Wx transposed [u][c], bf16

    const int tid  = threadIdx.x;
    const int lane = tid & 31;
    const int wave = tid >> 5;
    const long row0 = (long)blockIdx.x * 128;

    // stage Wx^T as bf16
    for (int e = tid; e < C_DIM * U_DIM; e += 256) {
        int c = e >> 8;      // e = c*256 + u
        int u = e & 255;
        lWxT[u * C_DIM + c] = f2bfbits(Wx[e]);
    }
    // stage A = BN1(x) tile as bf16
    for (int e = tid; e < 128 * C_DIM; e += 256) {
        int r = e >> 6;
        int c = e & 63;
        float inv = g1[c] * rsqrtf(v1[c] + 1e-3f);
        float sh  = b1[c] - m1[c] * inv;
        lA[e] = f2bfbits(x[(row0 + r) * C_DIM + c] * inv + sh);
    }
    __syncthreads();

    const int mrow  = wave * 16;            // this wave's 16-row M subtile
    const int am    = lane & 15;            // A: row within tile
    const int khalf = (lane >> 4) * 16;     // A: K-half per lane group
    const int kq    = (lane >> 4) * 16;     // B: K-half per lane group
    const int cm    = (lane >> 4) << 3;     // C: row base per lane group

    BFrag a0, a1;  // K = 0..31 and 32..63
    {
        const u16* pr = &lA[(mrow + am) * C_DIM];
        a0.h[0] = *(const v8u16*)(pr + 0  + khalf);
        a0.h[1] = *(const v8u16*)(pr + 8  + khalf);
        a1.h[0] = *(const v8u16*)(pr + 32 + khalf);
        a1.h[1] = *(const v8u16*)(pr + 40 + khalf);
    }

    for (int nt = 0; nt < 16; ++nt) {
        int u = nt * 16 + (lane & 15);
        BFrag b0, b1;
        const u16* pc = &lWxT[u * C_DIM];
        b0.h[0] = *(const v8u16*)(pc + kq + 0);
        b0.h[1] = *(const v8u16*)(pc + kq + 8);
        b1.h[0] = *(const v8u16*)(pc + 32 + kq + 0);
        b1.h[1] = *(const v8u16*)(pc + 32 + kq + 8);

        v8f acc = {};
        acc = wmma_bf16(a0, b0, acc);
        acc = wmma_bf16(a1, b1, acc);

        float bias = bh[u];
        #pragma unroll
        for (int r = 0; r < 8; ++r) {
            long grow = row0 + mrow + cm + r;   // = b*T + t
            long b = grow >> 10;
            long t = grow & 1023;
            xproj[(t * B_DIM + b) * U_DIM + u] = acc[r] + bias;
        }
    }
}

// ---------------------------------------------------------------------------
// Kernel 2: persistent RNN scan. 4 WGs x 512 threads (16 waves).
// WG g owns batch rows [16g,16g+16); wave w owns h columns [16w,16w+16).
// Wh bf16 fragments live in VGPRs; h state ping-pongs in LDS.
// xproj tile for step t+1 is prefetched into VGPRs during step t;
// xproj cachelines for step t+8 are prefetched into cache (global_prefetch).
// Two independent WMMA accumulator chains (k even / k odd).
// Fused BN2 on the way out; hn stored bf16 as [t][b][u].
// ---------------------------------------------------------------------------
__global__ __launch_bounds__(512) void k_rnn(
    const float* __restrict__ xproj, const float* __restrict__ Wh,
    const float* __restrict__ g2, const float* __restrict__ b2,
    const float* __restrict__ m2, const float* __restrict__ v2,
    u16* __restrict__ hn_out)
{
    __shared__ __align__(16) u16 h2[2][16 * U_DIM];   // ping-pong h state, bf16

    const int tid  = threadIdx.x;
    const int lane = tid & 31;
    const int wave = tid >> 5;           // 0..15
    const int b0   = blockIdx.x * 16;    // batch-row base
    const int nl   = lane & 15;
    const int u    = wave * 16 + nl;     // this lane's output column

    // h0 = 0
    for (int e = tid; e < 16 * U_DIM; e += 512) h2[0][e] = 0;

    // preload Wh B-fragments into VGPRs (one-time)
    const int kq = (lane >> 4) * 16;
    BFrag Bw[8];
    #pragma unroll
    for (int k = 0; k < 8; ++k)
        #pragma unroll
        for (int i = 0; i < 16; ++i)
            Bw[k].s[i] = f2bfbits(Wh[(k * 32 + kq + i) * U_DIM + u]);

    const float inv2 = g2[u] * rsqrtf(v2[u] + 1e-3f);
    const float sh2  = b2[u] - m2[u] * inv2;

    const int am    = lane & 15;
    const int khalf = (lane >> 4) * 16;
    const int cm    = (lane >> 4) << 3;
    __syncthreads();

    // prefetch xproj tile for t = 0
    v8f nxt;
    #pragma unroll
    for (int r = 0; r < 8; ++r)
        nxt[r] = xproj[(long)(b0 + cm + r) * U_DIM + u];

    for (int t = 0; t < T_DIM; ++t) {
        const int cur = t & 1, nx = cur ^ 1;

        v8f acc0 = nxt;     // xproj(t) already includes bh
        v8f acc1 = {};

        // L2/WGP$ prefetch 8 steps ahead (one cacheline per lane)
        if (t + 8 < T_DIM)
            __builtin_prefetch(xproj + ((long)(t + 8) * B_DIM + b0 + cm) * U_DIM + u, 0, 1);

        // VGPR prefetch xproj(t+1): latency hides behind WMMAs + tanh + barrier
        if (t + 1 < T_DIM) {
            const float* p = xproj + ((long)(t + 1) * B_DIM + b0) * U_DIM + u;
            #pragma unroll
            for (int r = 0; r < 8; ++r)
                nxt[r] = p[(long)(cm + r) * U_DIM];
        }

        // h @ Wh : two independent 4-deep WMMA chains
        #pragma unroll
        for (int k = 0; k < 8; ++k) {
            BFrag a;
            const u16* p = &h2[cur][am * U_DIM + k * 32 + khalf];
            a.h[0] = *(const v8u16*)p;
            a.h[1] = *(const v8u16*)(p + 8);
            if (k & 1) acc1 = wmma_bf16(a, Bw[k], acc1);
            else       acc0 = wmma_bf16(a, Bw[k], acc0);
        }

        // tanh, write next h state + fused BN2 output
        #pragma unroll
        for (int r = 0; r < 8; ++r) {
            float hv = tanhf(acc0[r] + acc1[r]);
            h2[nx][(cm + r) * U_DIM + u] = f2bfbits(hv);
            hn_out[((long)t * B_DIM + b0 + cm + r) * U_DIM + u] =
                f2bfbits(hv * inv2 + sh2);
        }
        __syncthreads();
    }
}

// ---------------------------------------------------------------------------
// Kernel 3: out = sigmoid(hn @ Wd + bd).  M=65536 (= T*B), K=256, N=20 (pad 32).
// hn rows are (t*64 + b); output is [b][t][o].
// 128 WGs x 256 threads; each wave does 4 M-tiles, both N-tiles.
// ---------------------------------------------------------------------------
__global__ __launch_bounds__(256) void k_dense(
    const u16* __restrict__ hn, const float* __restrict__ Wd,
    const float* __restrict__ bd, float* __restrict__ out)
{
    const int tid  = threadIdx.x;
    const int lane = tid & 31;
    const int wave = tid >> 5;          // 0..7
    const int nl   = lane & 15;
    const int kq   = (lane >> 4) * 16;

    // preload Wd fragments (pad columns >= 20 with zero)
    BFrag Wd0[8], Wd1[8];
    #pragma unroll
    for (int k = 0; k < 8; ++k)
        #pragma unroll
        for (int i = 0; i < 16; ++i) {
            int kk = k * 32 + kq + i;
            Wd0[k].s[i] = f2bfbits(Wd[kk * O_DIM + nl]);
            Wd1[k].s[i] = (16 + nl < O_DIM) ? f2bfbits(Wd[kk * O_DIM + 16 + nl]) : (u16)0;
        }
    const float bd0 = bd[nl];
    const float bd1 = (16 + nl < O_DIM) ? bd[16 + nl] : 0.f;

    const int am    = lane & 15;
    const int khalf = (lane >> 4) * 16;
    const int cm    = (lane >> 4) << 3;

    for (int i = 0; i < 4; ++i) {
        long tile = (long)blockIdx.x * 32 + wave * 4 + i;
        long row0 = tile * 16;
        v8f acc0 = {}, acc1 = {};
        #pragma unroll
        for (int k = 0; k < 8; ++k) {
            BFrag a;
            const u16* p = &hn[(row0 + am) * U_DIM + k * 32 + khalf];
            a.h[0] = *(const v8u16*)p;
            a.h[1] = *(const v8u16*)(p + 8);
            acc0 = wmma_bf16(a, Wd0[k], acc0);
            acc1 = wmma_bf16(a, Wd1[k], acc1);
        }
        #pragma unroll
        for (int r = 0; r < 8; ++r) {
            long grow = row0 + cm + r;          // = t*64 + b
            long b = grow & 63;
            long t = grow >> 6;
            float* o = out + (b * T_DIM + t) * O_DIM;
            o[nl] = 1.f / (1.f + __expf(-(acc0[r] + bd0)));
            if (16 + nl < O_DIM)
                o[16 + nl] = 1.f / (1.f + __expf(-(acc1[r] + bd1)));
        }
    }
}

// ---------------------------------------------------------------------------
extern "C" void kernel_launch(void* const* d_in, const int* in_sizes, int n_in,
                              void* d_out, int out_size, void* d_ws, size_t ws_size,
                              hipStream_t stream) {
    const float* x   = (const float*)d_in[0];
    const float* g1  = (const float*)d_in[1];
    const float* b1  = (const float*)d_in[2];
    const float* m1  = (const float*)d_in[3];
    const float* v1  = (const float*)d_in[4];
    const float* Wx  = (const float*)d_in[5];
    const float* Wh  = (const float*)d_in[6];
    const float* bh  = (const float*)d_in[7];
    const float* g2  = (const float*)d_in[8];
    const float* b2  = (const float*)d_in[9];
    const float* m2  = (const float*)d_in[10];
    const float* v2  = (const float*)d_in[11];
    const float* Wd  = (const float*)d_in[12];
    const float* bd  = (const float*)d_in[13];

    float* xproj = (float*)d_ws;                                   // 64 MiB, [t][b][u]
    u16*   hn    = (u16*)((char*)d_ws + (size_t)65536 * 256 * 4);  // 32 MiB, [t][b][u]

    k_bn1_xproj<<<512, 256, 0, stream>>>(x, g1, b1, m1, v1, Wx, bh, xproj);
    k_rnn<<<4, 512, 0, stream>>>(xproj, Wh, g2, b2, m2, v2, hn);
    k_dense<<<128, 256, 0, stream>>>(hn, Wd, bd, (float*)d_out);
}
